// GCN_58557584114028
// MI455X (gfx1250) — matrix operand order repaired and checked by reading, not compiled, not run on hardware
//
#include <hip/hip_runtime.h>
#include <hip/hip_bf16.h>

// ---------------------------------------------------------------------------
// 2-layer GCN for MI455X (gfx1250, wave32).
//  - GEMMs via v_wmma_f32_16x16x32_f16 (f16 inputs, f32 accumulate).
//  - Edge aggregation: float4 gather + global f32 atomics (L2-resident:
//    h1/out1 = 25.6 MB each << 192 MB L2).
//  - log_softmax: wave32 shuffle reductions.
// ---------------------------------------------------------------------------

typedef __attribute__((ext_vector_type(8)))  _Float16 v8h;
typedef __attribute__((ext_vector_type(16))) _Float16 v16h;
typedef __attribute__((ext_vector_type(8)))  float    v8f;

#define N_NODES 50000
#define F_IN    256
#define F_HID   128
#define F_OUT   64

// ---------------------------------------------------------------- degree ---
__global__ void gcn_deg_init(float* deg, int n) {
    int i = blockIdx.x * blockDim.x + threadIdx.x;
    if (i < n) deg[i] = 1.0f;               // self-loop contributes 1
}

__global__ void gcn_deg_count(const int* __restrict__ dst, float* deg, int E) {
    int e = blockIdx.x * blockDim.x + threadIdx.x;
    if (e < E) atomicAdd(&deg[dst[e]], 1.0f);
}

__global__ void gcn_rsqrt_inplace(float* deg, int n) {
    int i = blockIdx.x * blockDim.x + threadIdx.x;
    if (i < n) {
        float d = deg[i];
        deg[i] = (d > 0.0f) ? rsqrtf(d) : 0.0f;
    }
}

// ------------------------------------------------------------ WMMA GEMM ---
// H[M,N] = conv(X[M,K]) @ W[K,N], optional fused ReLU on X.
// One wave computes a 16x16 output tile; block has N/16 waves (full N per
// M-tile). W^T lives in LDS as f16 for contiguous per-column (B-fragment)
// reads; X tile converted f32->f16 into LDS per M-tile.
template <int K, int N, bool RELU>
__global__ __launch_bounds__(32 * (N / 16))
void gcn_gemm_wmma(const float* __restrict__ X, const float* __restrict__ W,
                   float* __restrict__ H, int M, int M_tiles)
{
    constexpr int PAD = 8;                  // halves of padding per LDS row
    constexpr int LDK = K + PAD;
    constexpr int NW  = N / 16;             // waves per block
    constexpr int NT  = 32 * NW;            // threads per block

    __shared__ _Float16 sW[N * LDK];        // W^T: [n][k], f16
    __shared__ _Float16 sA[16 * LDK];       // X tile: [r][k], f16

    const int tid = threadIdx.x;

    // Load W (row-major [K,N], coalesced) transposed into LDS as f16.
    for (int idx = tid; idx < K * N; idx += NT) {
        int k = idx / N, n = idx % N;
        sW[n * LDK + k] = (_Float16)W[idx];
    }
    __syncthreads();

    const int wave  = tid >> 5;
    const int lane  = tid & 31;
    const int row16 = lane & 15;            // M row / N col within tile
    const int hi    = lane >> 4;            // lane-half selector
    const int n0    = wave * 16;

    for (int mt = blockIdx.x; mt < M_tiles; mt += gridDim.x) {
        const bool full_tile = (mt * 16 + 16) <= M;   // uniform per block

        // --- stage X tile (16 rows x K) into LDS, f32 -> f16 (+ReLU) ---
        if (full_tile) {
            for (int idx = tid; idx < 16 * K; idx += NT) {
                int r = idx / K, k = idx % K;
                float v = X[(size_t)(mt * 16 + r) * K + k];
                if (RELU) v = v > 0.0f ? v : 0.0f;
                sA[r * LDK + k] = (_Float16)v;
            }
        } else {
            for (int idx = tid; idx < 16 * K; idx += NT) {
                int r = idx / K, k = idx % K;
                int grow = mt * 16 + r;
                float v = (grow < M) ? X[(size_t)grow * K + k] : 0.0f;
                if (RELU) v = v > 0.0f ? v : 0.0f;
                sA[r * LDK + k] = (_Float16)v;
            }
        }
        __syncthreads();

        v8f acc = {};
        #pragma unroll
        for (int kb = 0; kb < K; kb += 32) {
            // A fragment (16x32 f16): lane row = lane%16,
            //   halves 0..7  -> K = kb + hi*8 + 0..7
            //   halves 8..15 -> K = kb + hi*8 + 16..23
            const _Float16* ap = &sA[row16 * LDK + kb + hi * 8];
            v8h a_lo = *(const v8h*)ap;
            v8h a_hi = *(const v8h*)(ap + 16);
            v16h a = __builtin_shufflevector(a_lo, a_hi,
                        0,1,2,3,4,5,6,7,8,9,10,11,12,13,14,15);

            // B fragment (32x16 f16): lane col = n0 + lane%16,
            //   halves 0..7  -> K = kb + hi*8 + 0..7
            //   halves 8..15 -> K = kb + hi*8 + 16..23
            const _Float16* bp = &sW[(n0 + row16) * LDK + kb + hi * 8];
            v8h b_lo = *(const v8h*)bp;
            v8h b_hi = *(const v8h*)(bp + 16);
            v16h b = __builtin_shufflevector(b_lo, b_hi,
                        0,1,2,3,4,5,6,7,8,9,10,11,12,13,14,15);

            acc = __builtin_amdgcn_wmma_f32_16x16x32_f16(
                      false, a, false, b, (short)0, acc, false, false);
        }

        // C/D layout: lane holds N = lane%16, M = r + 8*hi for VGPR r.
        float* outp = H + ((size_t)mt * 16 + 8 * hi) * N + n0 + row16;
        if (full_tile) {                       // unguarded: EXEC stays all-1s
            #pragma unroll
            for (int r = 0; r < 8; ++r)
                outp[(size_t)r * N] = acc[r];
        } else {
            #pragma unroll
            for (int r = 0; r < 8; ++r) {
                int grow = mt * 16 + r + 8 * hi;
                if (grow < M) outp[(size_t)r * N] = acc[r];
            }
        }
        __syncthreads();
    }
}

// ----------------------------------------------- self-loop + bias init ----
// out[i][f] = b[f] + h[i][f] * dinv[i]^2     (self-loop edge, norm = dinv^2)
__global__ void gcn_self_bias(const float* __restrict__ h,
                              const float* __restrict__ dinv,
                              const float* __restrict__ b,
                              float* __restrict__ out, int n, int F)
{
    int idx = blockIdx.x * blockDim.x + threadIdx.x;
    if (idx >= n * F) return;
    int i = idx / F, f = idx % F;
    float di = dinv[i];
    out[idx] = b[f] + h[idx] * di * di;
}

// --------------------------------------------------------- edge scatter ---
// One lane per (edge, 4 features): gather float4 of h[src], scale by
// dinv[src]*dinv[dst], scatter-add into out[dst]. F/4 lanes per edge.
template <int F>
__global__ void gcn_edge_scatter(const float* __restrict__ h,
                                 const int* __restrict__ src,
                                 const int* __restrict__ dst,
                                 const float* __restrict__ dinv,
                                 float* __restrict__ out, int E)
{
    constexpr int LPE = F / 4;              // lanes per edge
    int tid = blockIdx.x * blockDim.x + threadIdx.x;
    int e = tid / LPE;
    if (e >= E) return;
    int c = (tid % LPE) * 4;

    int s = src[e], d = dst[e];
    float nrm = dinv[s] * dinv[d];
    float4 hv = *(const float4*)(h + (size_t)s * F + c);
    float* o = out + (size_t)d * F + c;
    atomicAdd(o + 0, hv.x * nrm);
    atomicAdd(o + 1, hv.y * nrm);
    atomicAdd(o + 2, hv.z * nrm);
    atomicAdd(o + 3, hv.w * nrm);
}

// ----------------------------------------------------- log_softmax (64) ---
// One wave32 per node, 2 features per lane, shuffle reductions.
__global__ void gcn_log_softmax64(float* __restrict__ out, int n)
{
    int node = blockIdx.x * (blockDim.x >> 5) + (threadIdx.x >> 5);
    int lane = threadIdx.x & 31;
    if (node >= n) return;

    float* row = out + (size_t)node * 64;
    float2 v = *(const float2*)(row + lane * 2);

    float m = fmaxf(v.x, v.y);
    #pragma unroll
    for (int off = 16; off > 0; off >>= 1)
        m = fmaxf(m, __shfl_xor(m, off, 32));

    float s = expf(v.x - m) + expf(v.y - m);
    #pragma unroll
    for (int off = 16; off > 0; off >>= 1)
        s += __shfl_xor(s, off, 32);

    float lse = m + logf(s);
    float2 r = {v.x - lse, v.y - lse};
    *(float2*)(row + lane * 2) = r;
}

// ------------------------------------------------------------- launcher ---
static inline int cdiv(int a, int b) { return (a + b - 1) / b; }

extern "C" void kernel_launch(void* const* d_in, const int* in_sizes, int n_in,
                              void* d_out, int out_size, void* d_ws, size_t ws_size,
                              hipStream_t stream)
{
    (void)n_in; (void)out_size; (void)ws_size;

    const float* x  = (const float*)d_in[0];   // [N, 256]
    const int*   ei = (const int*)  d_in[1];   // [2, E]
    const float* W1 = (const float*)d_in[2];   // [256, 128]
    const float* b1 = (const float*)d_in[3];   // [128]
    const float* W2 = (const float*)d_in[4];   // [128, 64]
    const float* b2 = (const float*)d_in[5];   // [64]
    float* out = (float*)d_out;                // [N, 64]

    const int Nn = in_sizes[0] / F_IN;         // 50000
    const int E  = in_sizes[1] / 2;            // 1600000
    const int* src = ei;
    const int* dst = ei + E;

    float* ws   = (float*)d_ws;
    float* dinv = ws;                                  // [Nn]
    float* h1   = dinv + Nn;                           // [Nn, 128]
    float* o1   = h1 + (size_t)Nn * F_HID;             // [Nn, 128]
    float* h2   = o1 + (size_t)Nn * F_HID;             // [Nn, 64]

    const int M_tiles = cdiv(Nn, 16);
    const int gemm_blocks = M_tiles < 960 ? M_tiles : 960;

    // degree -> dinv
    gcn_deg_init<<<cdiv(Nn, 256), 256, 0, stream>>>(dinv, Nn);
    gcn_deg_count<<<cdiv(E, 256), 256, 0, stream>>>(dst, dinv, E);
    gcn_rsqrt_inplace<<<cdiv(Nn, 256), 256, 0, stream>>>(dinv, Nn);

    // layer 1: h1 = x @ W1
    gcn_gemm_wmma<F_IN, F_HID, false>
        <<<gemm_blocks, 32 * (F_HID / 16), 0, stream>>>(x, W1, h1, Nn, M_tiles);
    // o1 = b1 + self-loop term, then scatter edges
    gcn_self_bias<<<cdiv(Nn * F_HID, 256), 256, 0, stream>>>(h1, dinv, b1, o1, Nn, F_HID);
    gcn_edge_scatter<F_HID>
        <<<cdiv(E * (F_HID / 4), 256), 256, 0, stream>>>(h1, src, dst, dinv, o1, E);

    // layer 2: h2 = relu(o1) @ W2   (ReLU fused into A-tile load)
    gcn_gemm_wmma<F_HID, F_OUT, true>
        <<<gemm_blocks, 32 * (F_OUT / 16), 0, stream>>>(o1, W2, h2, Nn, M_tiles);
    gcn_self_bias<<<cdiv(Nn * F_OUT, 256), 256, 0, stream>>>(h2, dinv, b2, out, Nn, F_OUT);
    gcn_edge_scatter<F_OUT>
        <<<cdiv(E * (F_OUT / 4), 256), 256, 0, stream>>>(h2, src, dst, dinv, out, E);

    // log_softmax over 64 features, in place on d_out
    gcn_log_softmax64<<<cdiv(Nn, 8), 256, 0, stream>>>(out, Nn);
}
